// Attention_36051955483052
// MI455X (gfx1250) — compile-verified, matrix operand order checked by reading, not run on hardware
//
#include <hip/hip_runtime.h>
#include <hip/hip_bf16.h>

// MI455X (gfx1250) attention block:
//   cast fp32->f16  ->  WMMA GEMM QKV (+fused RoPE, V transposed)
//   ->  per-wave flash attention (WMMA S=QK^T, online softmax, WMMA O+=PV,
//       Q tile staged in LDS to kill register spills)
//   ->  WMMA GEMM output projection (fp32 out)
// Workspace layout (needs 112 MB):
//   [0,16M)   Xh    f16 [4096,2048]
//   [16,24M)  Wqh   f16 [2048,2048]
//   [24,32M)  Wkh
//   [32,40M)  Wvh
//   [40,48M)  Woh
//   [48,64M)  Qh    f16 [B,H,T,HD]
//   [64,80M)  Kh    f16 [B,H,T,HD]
//   [80,96M)  Vt    f16 [B,H,HD,T]   (transposed for contiguous PV B-frags)
//   [96,112M) Oh    f16 [B,T,H*HD]

typedef __attribute__((ext_vector_type(16))) _Float16 v16h;
typedef __attribute__((ext_vector_type(8)))  _Float16 v8h;
typedef __attribute__((ext_vector_type(8)))  float    v8f;

#define B_  2
#define T_  2048
#define D_  2048
#define H_  16
#define HD_ 128
#define M_  (B_ * T_)   // 4096

enum { MODE_ROPE = 0, MODE_VT = 1, MODE_F32 = 2 };

static __device__ __forceinline__ v8f vzero8() {
  v8f v;
#pragma unroll
  for (int i = 0; i < 8; ++i) v[i] = 0.0f;
  return v;
}

// Load one 16-half WMMA fragment (A- or B-matrix) for this lane.
// rowPtr = start (k=0) of this lane's row/column; kOff = kBase + 8*hi.
// Layout (ISA 7.12.2, 16-bit A 16x32): elems 0..7 -> K=kOff..kOff+7,
// elems 8..15 -> K=kOff+16..kOff+23.
static __device__ __forceinline__ v16h load_frag(const _Float16* rowPtr, int kOff) {
  union { v16h v; struct { v8h lo; v8h hi; } s; } u;
  u.s.lo = *(const v8h*)(rowPtr + kOff);
  u.s.hi = *(const v8h*)(rowPtr + kOff + 16);
  return u.v;
}

static __device__ __forceinline__ v8f wmma_f16(v16h a, v16h b, v8f c) {
  return __builtin_amdgcn_wmma_f32_16x16x32_f16(false, a, false, b, (short)0, c,
                                                false, false);
}

// ---------------------------------------------------------------- cast kernel
__global__ void k_cast_f32_f16(const float* __restrict__ src,
                               _Float16* __restrict__ dst, int n) {
  int i = blockIdx.x * blockDim.x + threadIdx.x;
  int stride = gridDim.x * blockDim.x;
  for (; i < n; i += stride) dst[i] = (_Float16)src[i];
}

// ------------------------------------------------------------- WMMA GEMM
// Y = A[M_,D_] @ W[D_,D_]^T  ;  block tile 128x128, wave tile 32x64.
template <int MODE>
__global__ __launch_bounds__(256, 1) void k_gemm_xwT(
    const _Float16* __restrict__ A, const _Float16* __restrict__ W,
    void* __restrict__ dstv, const float* __restrict__ rope) {
  const int lane = threadIdx.x & 31;
  const int wid  = threadIdx.x >> 5;
  const int hi   = lane >> 4;
  const int ln16 = lane & 15;
  const int mBase = blockIdx.y * 128 + (wid & 3) * 32;
  const int nBase = blockIdx.x * 128 + (wid >> 2) * 64;

  v8f acc[2][4];
#pragma unroll
  for (int mi = 0; mi < 2; ++mi)
#pragma unroll
    for (int ni = 0; ni < 4; ++ni) acc[mi][ni] = vzero8();

  const _Float16* aRow0 = A + (size_t)(mBase + ln16) * D_;
  const _Float16* aRow1 = aRow0 + (size_t)16 * D_;
  const _Float16* bRow[4];
#pragma unroll
  for (int ni = 0; ni < 4; ++ni)
    bRow[ni] = W + (size_t)(nBase + ni * 16 + ln16) * D_;

  for (int kb = 0; kb < D_; kb += 32) {
    const int kOff = kb + 8 * hi;
    v16h a0 = load_frag(aRow0, kOff);
    v16h a1 = load_frag(aRow1, kOff);
#pragma unroll
    for (int ni = 0; ni < 4; ++ni) {
      v16h b = load_frag(bRow[ni], kOff);
      acc[0][ni] = wmma_f16(a0, b, acc[0][ni]);
      acc[1][ni] = wmma_f16(a1, b, acc[1][ni]);
    }
  }

  // Epilogue. D layout: element i of lane L -> row i + 8*(L>>4), col L&15.
#pragma unroll
  for (int mi = 0; mi < 2; ++mi) {
#pragma unroll
    for (int ni = 0; ni < 4; ++ni) {
#pragma unroll
      for (int i = 0; i < 8; ++i) {
        const int m = mBase + mi * 16 + i + 8 * hi;
        const int n = nBase + ni * 16 + ln16;
        float v = acc[mi][ni][i];
        if (MODE == MODE_F32) {
          ((float*)dstv)[(size_t)m * D_ + n] = v;
        } else {
          const int b  = m >> 11;        // m / T_
          const int t  = m & (T_ - 1);
          const int h  = n >> 7;         // n / HD_
          const int hd = n & (HD_ - 1);
          if (MODE == MODE_ROPE) {
            // RoPE pair (2j, 2j+1) sits in the lane-neighbor (cols stripe lanes).
            float p = __shfl_xor(v, 1, 32);
            const int j = hd >> 1;
            float c = rope[((size_t)t * (HD_ / 2) + j) * 2 + 0];
            float s = rope[((size_t)t * (HD_ / 2) + j) * 2 + 1];
            float r = (hd & 1) ? (p * s + v * c) : (v * c - p * s);
            ((_Float16*)dstv)[(((size_t)(b * H_ + h)) * T_ + t) * HD_ + hd] =
                (_Float16)r;
          } else {  // MODE_VT : V stored [B,H,HD,T]
            ((_Float16*)dstv)[(((size_t)(b * H_ + h)) * HD_ + hd) * T_ + t] =
                (_Float16)v;
          }
        }
      }
    }
  }
  (void)rope;
}

// ------------------------------------------------------------ flash attention
// Each wave: one 16-query tile of one (b,h); online softmax over 32-key blocks.
// Q tile lives in LDS (4 KB/wave) so only transient fragments occupy VGPRs.
__global__ __launch_bounds__(256, 1) void k_flash_attn(
    const _Float16* __restrict__ Qh, const _Float16* __restrict__ Kh,
    const _Float16* __restrict__ Vt, _Float16* __restrict__ Oh) {
  __shared__ alignas(16) _Float16 qtile[8][16 * HD_];  // 32 KB: per-wave Q tile
  __shared__ alignas(16) _Float16 pbuf[8][16 * 32];    //  8 KB: P transpose buf

  const int lane = threadIdx.x & 31;
  const int wid  = threadIdx.x >> 5;
  const int hi   = lane >> 4;
  const int ln16 = lane & 15;

  const int wtile = blockIdx.x * 8 + wid;
  const int q16 = wtile & (T_ / 16 - 1);  // 128 q-tiles per (b,h)
  const int bh  = wtile >> 7;
  const int q0  = q16 * 16;

  const size_t baseQK = (size_t)bh * T_ * HD_;
  const _Float16* qRow = Qh + baseQK + (size_t)(q0 + ln16) * HD_;
  const _Float16* kRow = Kh + baseQK + (size_t)ln16 * HD_;
  const _Float16* vCol = Vt + (size_t)bh * HD_ * T_ + (size_t)ln16 * T_;
  _Float16* qLds = &qtile[wid][0];

  // Copy 16x128 Q tile to LDS: lane handles row ln16, half `hi` (128 B each).
#pragma unroll
  for (int c = 0; c < 8; ++c) {
    *(v8h*)&qLds[ln16 * HD_ + hi * 64 + c * 8] =
        *(const v8h*)(qRow + hi * 64 + c * 8);
  }
  asm volatile("s_wait_dscnt 0" ::: "memory");

  v8f o[8];
#pragma unroll
  for (int nt = 0; nt < 8; ++nt) o[nt] = vzero8();
  float mstat[8], lstat[8];
#pragma unroll
  for (int i = 0; i < 8; ++i) { mstat[i] = -3.0e38f; lstat[i] = 0.0f; }

  const float scale = 0.08838834764831845f;  // 1/sqrt(128)

  for (int kb = 0; kb < q0 + 16; kb += 32) {
    // ---- S = Q @ K^T for keys [kb, kb+32); Q fragments streamed from LDS
    v8f s0 = vzero8(), s1 = vzero8();
#pragma unroll
    for (int c = 0; c < 4; ++c) {
      v16h qf = load_frag(qLds + ln16 * HD_, c * 32 + 8 * hi);
      v16h k0 = load_frag(kRow + (size_t)kb * HD_,        c * 32 + 8 * hi);
      v16h k1 = load_frag(kRow + (size_t)(kb + 16) * HD_, c * 32 + 8 * hi);
      s0 = wmma_f16(qf, k0, s0);
      s1 = wmma_f16(qf, k1, s1);
    }

    // ---- causal mask + online softmax (rows = element idx, cols = lanes)
#pragma unroll
    for (int i = 0; i < 8; ++i) {
      const int tq = q0 + i + 8 * hi;
      float e0 = (kb + ln16      <= tq) ? s0[i] * scale : -3.0e38f;
      float e1 = (kb + 16 + ln16 <= tq) ? s1[i] * scale : -3.0e38f;
      float rm = fmaxf(e0, e1);
#pragma unroll
      for (int off = 1; off <= 8; off <<= 1)
        rm = fmaxf(rm, __shfl_xor(rm, off, 32));
      const float mnew  = fmaxf(mstat[i], rm);
      const float alpha = __expf(mstat[i] - mnew);
      const float p0 = __expf(e0 - mnew);
      const float p1 = __expf(e1 - mnew);
      float rs = p0 + p1;
#pragma unroll
      for (int off = 1; off <= 8; off <<= 1) rs += __shfl_xor(rs, off, 32);
      lstat[i] = lstat[i] * alpha + rs;
      mstat[i] = mnew;
#pragma unroll
      for (int nt = 0; nt < 8; ++nt) o[nt][i] *= alpha;
      // stash P row-major 16x32 in per-wave LDS (D-layout -> A-layout)
      const int row = i + 8 * hi;
      pbuf[wid][row * 32 + ln16]      = (_Float16)p0;
      pbuf[wid][row * 32 + 16 + ln16] = (_Float16)p1;
    }
    asm volatile("s_wait_dscnt 0" ::: "memory");

    // ---- read P back as A-fragment
    union { v16h v; struct { v8h lo; v8h hi; } s; } pu;
    pu.s.lo = *(const v8h*)&pbuf[wid][ln16 * 32 + 8 * hi];
    pu.s.hi = *(const v8h*)&pbuf[wid][ln16 * 32 + 16 + 8 * hi];
    asm volatile("" ::: "memory");  // keep next iter's DS stores after these loads

    // ---- O += P @ V  (V transposed -> contiguous B-fragment loads)
#pragma unroll
    for (int nt = 0; nt < 8; ++nt) {
      v16h vf = load_frag(vCol + (size_t)nt * 16 * T_, kb + 8 * hi);
      o[nt] = wmma_f16(pu.v, vf, o[nt]);
    }
  }

  // ---- normalize and store O as [B, T, H*HD] f16
  const int b = bh >> 4;
  const int h = bh & 15;
#pragma unroll
  for (int nt = 0; nt < 8; ++nt) {
#pragma unroll
    for (int i = 0; i < 8; ++i) {
      const int t  = q0 + i + 8 * hi;
      const int hd = nt * 16 + ln16;
      float val = o[nt][i] / lstat[i];
      Oh[(((size_t)(b * T_ + t)) * H_ + h) * HD_ + hd] = (_Float16)val;
    }
  }
}

// ---------------------------------------------------------------- launcher
extern "C" void kernel_launch(void* const* d_in, const int* in_sizes, int n_in,
                              void* d_out, int out_size, void* d_ws,
                              size_t ws_size, hipStream_t stream) {
  const float* x    = (const float*)d_in[0];
  const float* rope = (const float*)d_in[1];
  const float* w_q  = (const float*)d_in[2];
  const float* w_k  = (const float*)d_in[3];
  const float* w_v  = (const float*)d_in[4];
  const float* w_o  = (const float*)d_in[5];
  float* out = (float*)d_out;

  char* ws = (char*)d_ws;
  const size_t MB = 1u << 20;
  _Float16* Xh  = (_Float16*)(ws + 0 * MB);
  _Float16* Wqh = (_Float16*)(ws + 16 * MB);
  _Float16* Wkh = (_Float16*)(ws + 24 * MB);
  _Float16* Wvh = (_Float16*)(ws + 32 * MB);
  _Float16* Woh = (_Float16*)(ws + 40 * MB);
  _Float16* Qh  = (_Float16*)(ws + 48 * MB);
  _Float16* Kh  = (_Float16*)(ws + 64 * MB);
  _Float16* Vt  = (_Float16*)(ws + 80 * MB);
  _Float16* Oh  = (_Float16*)(ws + 96 * MB);

  k_cast_f32_f16<<<2048, 256, 0, stream>>>(x, Xh, M_ * D_);
  k_cast_f32_f16<<<1024, 256, 0, stream>>>(w_q, Wqh, D_ * D_);
  k_cast_f32_f16<<<1024, 256, 0, stream>>>(w_k, Wkh, D_ * D_);
  k_cast_f32_f16<<<1024, 256, 0, stream>>>(w_v, Wvh, D_ * D_);
  k_cast_f32_f16<<<1024, 256, 0, stream>>>(w_o, Woh, D_ * D_);

  dim3 gg(D_ / 128, M_ / 128);  // (16, 32), 256 threads
  k_gemm_xwT<MODE_ROPE><<<gg, 256, 0, stream>>>(Xh, Wqh, Qh, rope);
  k_gemm_xwT<MODE_ROPE><<<gg, 256, 0, stream>>>(Xh, Wkh, Kh, rope);
  k_gemm_xwT<MODE_VT><<<gg, 256, 0, stream>>>(Xh, Wvh, Vt, nullptr);

  k_flash_attn<<<(B_ * H_ * (T_ / 16)) / 8, 256, 0, stream>>>(Qh, Kh, Vt, Oh);

  k_gemm_xwT<MODE_F32><<<gg, 256, 0, stream>>>(Oh, Woh, out, nullptr);

  (void)in_sizes; (void)n_in; (void)out_size; (void)ws_size;
}
